// EmbedMatcher_48816598286877
// MI455X (gfx1250) — compile-verified, matrix operand order checked by reading, not run on hardware
//
#include <hip/hip_runtime.h>
#include <hip/hip_bf16.h>
#include <math.h>

// ---------------------------------------------------------------------------
// EmbedMatcher for MI455X (gfx1250).  wave32, WMMA bf16 -> f32 accumulate.
// Dims: Bq=4096, Bs=5, M=32, D=256, DM=512, H=1024, 4H=4096, STEPS=4
// GEMM: one wave computes a 32(m) x 64(n) tile -> 8 v_wmma per 32-k step,
// 12 b128 loads per step (vs 10:4 before) => half the B-operand bytes/FLOP.
// ---------------------------------------------------------------------------

typedef __attribute__((ext_vector_type(16))) __bf16 v16bf;
typedef __attribute__((ext_vector_type(8)))  __bf16 v8bf;
typedef __attribute__((ext_vector_type(8)))  float  v8f;

#define BQ 4096
#define BS 5
#define MM 32
#define DD 256
#define DM 512
#define HH 1024
#define FH 4096

// ---------------- gather-sum over neighbors -> S[b, 0:256]=sum rel, S[b,256:512]=sum ent
__global__ void gather_sum_kernel(const int* __restrict__ conn,
                                  const float* __restrict__ emb,
                                  __bf16* __restrict__ outb,
                                  float* __restrict__ outf, int B) {
  int gid = blockIdx.x * blockDim.x + threadIdx.x;
  if (gid >= B * DM) return;
  int b = gid >> 9, d = gid & 511;
  int col = d >> 8;          // 0 = relation, 1 = entity
  int dd = d & 255;
  const int* cb = conn + b * (MM * 2);
  float acc = 0.f;
#pragma unroll 4
  for (int m = 0; m < MM; ++m) {
    long long idx = (long long)cb[m * 2 + col];
    acc += emb[idx * DD + dd];
  }
  if (outb) outb[gid] = (__bf16)acc;
  if (outf) outf[gid] = acc;
}

// ---------------- f32 -> bf16 (strided source) for weight conversion
__global__ void convert_bf16_kernel(const float* __restrict__ src,
                                    __bf16* __restrict__ dst,
                                    int rows, int cols, int src_stride, int src_off) {
  int gid = blockIdx.x * blockDim.x + threadIdx.x;
  if (gid >= rows * cols) return;
  int r = gid / cols, c = gid - r * cols;
  dst[gid] = (__bf16)src[(long long)r * src_stride + src_off + c];
}

// ---------------- per-lane 16xK tile fetch (A: [m,k] row-major; B: W[n,k] row-major)
// lane -> r=lane%16 (m or n), half=lane/16; elems 0..7 = k0+half*8+.., 8..15 = k0+16+half*8+..
__device__ __forceinline__ v16bf load_frag(const __bf16* base) {
  v8bf lo = *(const v8bf*)(base);
  v8bf hi = *(const v8bf*)(base + 16);
  return __builtin_shufflevector(lo, hi, 0, 1, 2, 3, 4, 5, 6, 7,
                                 8, 9, 10, 11, 12, 13, 14, 15);
}

// ---------------- C[M,N] = A[M,K] * W[N,K]^T ; wave computes 32(m) x 64(n)
__global__ void wmma_gemm_bf16(const __bf16* __restrict__ A,
                               const __bf16* __restrict__ W,
                               float* __restrict__ C, int Mr, int Nr, int K) {
  int wave = threadIdx.x >> 5;
  int lane = threadIdx.x & 31;
  int r = lane & 15, half = lane >> 4;
  int tileM = blockIdx.x;                 // 32-row tiles
  int tileN = blockIdx.y * 8 + wave;      // 64-col tiles
  if (tileN * 64 >= Nr) return;

  const __bf16* arow0 = A + (long long)(tileM * 32 + r) * K + half * 8;
  const __bf16* arow1 = arow0 + (long long)16 * K;
  const __bf16* b0 = W + (long long)(tileN * 64 + r) * K + half * 8;
  const __bf16* b1 = b0 + (long long)16 * K;
  const __bf16* b2 = b0 + (long long)32 * K;
  const __bf16* b3 = b0 + (long long)48 * K;

  v8f z = {0.f, 0.f, 0.f, 0.f, 0.f, 0.f, 0.f, 0.f};
  v8f acc00 = z, acc01 = z, acc02 = z, acc03 = z;
  v8f acc10 = z, acc11 = z, acc12 = z, acc13 = z;

  for (int k0 = 0; k0 < K; k0 += 32) {
    __builtin_prefetch(b0 + k0 + 128, 0, 3);   // pull B stream into near cache
    v16bf a0 = load_frag(arow0 + k0);
    v16bf a1 = load_frag(arow1 + k0);
    v16bf f0 = load_frag(b0 + k0);
    v16bf f1 = load_frag(b1 + k0);
    v16bf f2 = load_frag(b2 + k0);
    v16bf f3 = load_frag(b3 + k0);
    acc00 = __builtin_amdgcn_wmma_f32_16x16x32_bf16(false, a0, false, f0, (short)0, acc00, false, false);
    acc01 = __builtin_amdgcn_wmma_f32_16x16x32_bf16(false, a0, false, f1, (short)0, acc01, false, false);
    acc02 = __builtin_amdgcn_wmma_f32_16x16x32_bf16(false, a0, false, f2, (short)0, acc02, false, false);
    acc03 = __builtin_amdgcn_wmma_f32_16x16x32_bf16(false, a0, false, f3, (short)0, acc03, false, false);
    acc10 = __builtin_amdgcn_wmma_f32_16x16x32_bf16(false, a1, false, f0, (short)0, acc10, false, false);
    acc11 = __builtin_amdgcn_wmma_f32_16x16x32_bf16(false, a1, false, f1, (short)0, acc11, false, false);
    acc12 = __builtin_amdgcn_wmma_f32_16x16x32_bf16(false, a1, false, f2, (short)0, acc12, false, false);
    acc13 = __builtin_amdgcn_wmma_f32_16x16x32_bf16(false, a1, false, f3, (short)0, acc13, false, false);
  }
#pragma unroll
  for (int i = 0; i < 8; ++i) {
    long long m0 = tileM * 32 + i + 8 * half;
    float* c0 = C + m0 * (long long)Nr + tileN * 64 + r;
    c0[0] = acc00[i];  c0[16] = acc01[i];  c0[32] = acc02[i];  c0[48] = acc03[i];
    float* c1 = c0 + (long long)16 * Nr;
    c1[0] = acc10[i];  c1[16] = acc11[i];  c1[32] = acc12[i];  c1[48] = acc13[i];
  }
}

// ---------------- tanh((C + 32*(gwb+gb))/deg) -> QN (f32 + bf16), side selects halves
__global__ void neighbor_epilogue(const float* __restrict__ Cq, const float* __restrict__ deg,
                                  const float* __restrict__ gwb, const float* __restrict__ gb,
                                  float* __restrict__ QNf, __bf16* __restrict__ QNb,
                                  int B, int side) {
  int gid = blockIdx.x * blockDim.x + threadIdx.x;
  if (gid >= B * DD) return;
  int b = gid >> 8, d = gid & 255;
  float v = tanhf((Cq[gid] + 32.f * (gwb[d] + gb[d])) / deg[b]);
  int o = b * DM + side * DD + d;
  QNf[o] = v;
  QNb[o] = (__bf16)v;
}

// ---------------- support: neighbor encode (f32, tiny)
__global__ void support_neighbor_kernel(const float* __restrict__ Sl, const float* __restrict__ Sr,
                                        const float* __restrict__ gw,
                                        const float* __restrict__ gwb, const float* __restrict__ gb,
                                        const float* __restrict__ degl, const float* __restrict__ degr,
                                        float* __restrict__ sn) {
  int gid = blockIdx.x * blockDim.x + threadIdx.x;
  if (gid >= BS * DM) return;
  int b = gid / DM, d = gid - b * DM;
  const float* S = (d < DD) ? (Sl + b * DM) : (Sr + b * DM);
  int row = d & 255;
  const float* w = gw + row * DM;
  float acc = 0.f;
  for (int k = 0; k < DM; ++k) acc += S[k] * w[k];
  float dg = (d < DD) ? degl[b] : degr[b];
  sn[gid] = tanhf((acc + 32.f * (gwb[row] + gb[row])) / dg);
}

// ---------------- support MLP layer 1: h1 = relu(sn @ p1_w.T + p1_b)  [5,1024]
__global__ void support_mlp1_kernel(const float* __restrict__ sn, const float* __restrict__ p1w,
                                    const float* __restrict__ p1b, float* __restrict__ h1) {
  int gid = blockIdx.x * blockDim.x + threadIdx.x;
  if (gid >= BS * HH) return;
  int b = gid / HH, j = gid - b * HH;
  const float* x = sn + b * DM;
  const float* w = p1w + (long long)j * DM;
  float acc = p1b[j];
  for (int k = 0; k < DM; ++k) acc += x[k] * w[k];
  h1[gid] = acc > 0.f ? acc : 0.f;
}

// ---------------- support MLP layer 2 + residual: y = h1 @ p2_w.T + p2_b + sn  [5,512]
__global__ void support_mlp2_kernel(const float* __restrict__ h1, const float* __restrict__ p2w,
                                    const float* __restrict__ p2b, const float* __restrict__ sn,
                                    float* __restrict__ y) {
  int gid = blockIdx.x * blockDim.x + threadIdx.x;
  if (gid >= BS * DM) return;
  int b = gid / DM, d = gid - b * DM;
  const float* x = h1 + b * HH;
  const float* w = p2w + (long long)d * HH;
  float acc = p2b[d];
  for (int k = 0; k < HH; ++k) acc += x[k] * w[k];
  y[gid] = acc + sn[gid];
}

// ---------------- LayerNorm (unbiased std, eps added to sigma) + mean over 5 rows -> sg[512]
__global__ void support_ln_mean_kernel(const float* __restrict__ y, const float* __restrict__ ln_g,
                                       const float* __restrict__ ln_b, float* __restrict__ sg) {
  __shared__ float red[DM];
  int d = threadIdx.x;
  float acc = 0.f;
  for (int b = 0; b < BS; ++b) {
    float v = y[b * DM + d];
    red[d] = v;
    __syncthreads();
    for (int s = DM / 2; s > 0; s >>= 1) {
      if (d < s) red[d] += red[d + s];
      __syncthreads();
    }
    float mu = red[0] * (1.f / DM);
    __syncthreads();
    float dv = v - mu;
    red[d] = dv * dv;
    __syncthreads();
    for (int s = DM / 2; s > 0; s >>= 1) {
      if (d < s) red[d] += red[d + s];
      __syncthreads();
    }
    float sigma = sqrtf(red[0] / (float)(DM - 1));
    __syncthreads();
    acc += ln_g[d] * dv / (sigma + 1e-3f) + ln_b[d];
  }
  sg[d] = acc * (1.f / BS);
}

// ---------------- constants: bias0 = b_ih + b_hh ; gc[n] = sg . w_hh[n, 512:1024]
__global__ void bias0_kernel(const float* __restrict__ b_ih, const float* __restrict__ b_hh,
                             float* __restrict__ bias0) {
  int n = blockIdx.x * blockDim.x + threadIdx.x;
  if (n < FH) bias0[n] = b_ih[n] + b_hh[n];
}
__global__ void gconst_kernel(const float* __restrict__ w_hh, const float* __restrict__ sg,
                              float* __restrict__ gc) {
  int n = blockIdx.x * blockDim.x + threadIdx.x;
  if (n >= FH) return;
  const float* w = w_hh + (long long)n * HH + DM;
  float acc = 0.f;
  for (int k = 0; k < DM; ++k) acc += sg[k] * w[k];
  gc[n] = acc;
}

// ---------------- fused LSTM gate nonlinearity.  gates = qx + bias0 [+ G + gc]
__global__ void lstm_step_kernel(const float* __restrict__ qx, const float* __restrict__ G,
                                 const float* __restrict__ bias0, const float* __restrict__ gc,
                                 float* __restrict__ c, const float* __restrict__ QNf,
                                 float* __restrict__ hf, __bf16* __restrict__ hb, int useG) {
  int gid = blockIdx.x * blockDim.x + threadIdx.x;
  if (gid >= BQ * HH) return;
  int b = gid >> 10, j = gid & 1023;
  long long base = (long long)b * FH;
  float gi = qx[base + j] + bias0[j];
  float gf = qx[base + HH + j] + bias0[HH + j];
  float gg = qx[base + 2 * HH + j] + bias0[2 * HH + j];
  float go = qx[base + 3 * HH + j] + bias0[3 * HH + j];
  float cold = 0.f;
  if (useG) {
    gi += G[base + j] + gc[j];
    gf += G[base + HH + j] + gc[HH + j];
    gg += G[base + 2 * HH + j] + gc[2 * HH + j];
    go += G[base + 3 * HH + j] + gc[3 * HH + j];
    cold = c[gid];
  }
  float si = 1.f / (1.f + expf(-gi));
  float sf = 1.f / (1.f + expf(-gf));
  float so = 1.f / (1.f + expf(-go));
  float cn = sf * cold + si * tanhf(gg);
  c[gid] = cn;
  float hc = so * tanhf(cn);
  if (j < DM) {
    float hv = QNf[b * DM + j] + hc;
    hf[b * DM + j] = hv;
    hb[b * DM + j] = (__bf16)hv;
  }
}

// ---------------- out[b] = h[b] . sg   (one wave32 per row)
__global__ void final_dot_kernel(const float* __restrict__ hf, const float* __restrict__ sg,
                                 float* __restrict__ out) {
  int wave = threadIdx.x >> 5, lane = threadIdx.x & 31;
  int b = blockIdx.x * 8 + wave;
  if (b >= BQ) return;
  const float* row = hf + b * DM;
  float acc = 0.f;
  for (int d = lane; d < DM; d += 32) acc += row[d] * sg[d];
  for (int off = 16; off > 0; off >>= 1) acc += __shfl_down(acc, off, 32);
  if (lane == 0) out[b] = acc;
}

// ===========================================================================
extern "C" void kernel_launch(void* const* d_in, const int* in_sizes, int n_in,
                              void* d_out, int out_size, void* d_ws, size_t ws_size,
                              hipStream_t stream) {
  const int*   q_left_conn  = (const int*)d_in[0];
  const float* q_left_deg   = (const float*)d_in[1];
  const int*   q_right_conn = (const int*)d_in[2];
  const float* q_right_deg  = (const float*)d_in[3];
  const int*   s_left_conn  = (const int*)d_in[4];
  const float* s_left_deg   = (const float*)d_in[5];
  const int*   s_right_conn = (const int*)d_in[6];
  const float* s_right_deg  = (const float*)d_in[7];
  const float* symbol_emb   = (const float*)d_in[8];
  const float* gcn_w        = (const float*)d_in[9];
  const float* gcn_w_b      = (const float*)d_in[10];
  const float* gcn_b        = (const float*)d_in[11];
  const float* p1_w         = (const float*)d_in[12];
  const float* p1_b         = (const float*)d_in[13];
  const float* p2_w         = (const float*)d_in[14];
  const float* p2_b         = (const float*)d_in[15];
  const float* ln_g         = (const float*)d_in[16];
  const float* ln_b         = (const float*)d_in[17];
  const float* w_ih         = (const float*)d_in[18];
  const float* w_hh         = (const float*)d_in[19];
  const float* b_ih         = (const float*)d_in[20];
  const float* b_hh         = (const float*)d_in[21];
  float* out = (float*)d_out;

  // ---- workspace carve-out (~200 MB)
  char* ws = (char*)d_ws;
  size_t off = 0;
  auto carve = [&](size_t bytes) -> void* {
    off = (off + 255) & ~(size_t)255;
    void* p = ws + off;
    off += bytes;
    return p;
  };
  __bf16* Sql_b = (__bf16*)carve((size_t)BQ * DM * 2);
  __bf16* Sqr_b = (__bf16*)carve((size_t)BQ * DM * 2);
  float*  Ssl_f = (float*)carve((size_t)BS * DM * 4);
  float*  Ssr_f = (float*)carve((size_t)BS * DM * 4);
  __bf16* gw_b  = (__bf16*)carve((size_t)DD * DM * 2);
  __bf16* wih_b = (__bf16*)carve((size_t)FH * DM * 2);
  __bf16* whh_b = (__bf16*)carve((size_t)FH * DM * 2);
  float*  Cql   = (float*)carve((size_t)BQ * DD * 4);
  float*  Cqr   = (float*)carve((size_t)BQ * DD * 4);
  float*  QNf   = (float*)carve((size_t)BQ * DM * 4);
  __bf16* QNb   = (__bf16*)carve((size_t)BQ * DM * 2);
  float*  sn    = (float*)carve((size_t)BS * DM * 4);
  float*  h1    = (float*)carve((size_t)BS * HH * 4);
  float*  ybuf  = (float*)carve((size_t)BS * DM * 4);
  float*  sg    = (float*)carve((size_t)DM * 4);
  float*  bias0 = (float*)carve((size_t)FH * 4);
  float*  gc    = (float*)carve((size_t)FH * 4);
  float*  qx    = (float*)carve((size_t)BQ * FH * 4);
  float*  Gm    = (float*)carve((size_t)BQ * FH * 4);
  float*  cbuf  = (float*)carve((size_t)BQ * HH * 4);
  float*  hf    = (float*)carve((size_t)BQ * DM * 4);
  __bf16* hb    = (__bf16*)carve((size_t)BQ * DM * 2);
  (void)ws_size; (void)in_sizes; (void)n_in; (void)out_size;

  const int T = 256;
  // 1) neighbor gather-sums (queries -> bf16 for WMMA, support -> f32)
  gather_sum_kernel<<<(BQ * DM + T - 1) / T, T, 0, stream>>>(q_left_conn, symbol_emb, Sql_b, nullptr, BQ);
  gather_sum_kernel<<<(BQ * DM + T - 1) / T, T, 0, stream>>>(q_right_conn, symbol_emb, Sqr_b, nullptr, BQ);
  gather_sum_kernel<<<(BS * DM + T - 1) / T, T, 0, stream>>>(s_left_conn, symbol_emb, nullptr, Ssl_f, BS);
  gather_sum_kernel<<<(BS * DM + T - 1) / T, T, 0, stream>>>(s_right_conn, symbol_emb, nullptr, Ssr_f, BS);

  // 2) weight conversion to bf16
  convert_bf16_kernel<<<(DD * DM + T - 1) / T, T, 0, stream>>>(gcn_w, gw_b, DD, DM, DM, 0);
  convert_bf16_kernel<<<(FH * DM + T - 1) / T, T, 0, stream>>>(w_ih, wih_b, FH, DM, DM, 0);
  convert_bf16_kernel<<<(FH * DM + T - 1) / T, T, 0, stream>>>(w_hh, whh_b, FH, DM, HH, 0);

  // 3) query neighbor GEMMs  [4096,512] x [256,512]^T -> [4096,256]
  {
    dim3 grid(BQ / 32, ((DD / 64) + 7) / 8);
    wmma_gemm_bf16<<<grid, 256, 0, stream>>>(Sql_b, gw_b, Cql, BQ, DD, DM);
    wmma_gemm_bf16<<<grid, 256, 0, stream>>>(Sqr_b, gw_b, Cqr, BQ, DD, DM);
  }
  neighbor_epilogue<<<(BQ * DD + T - 1) / T, T, 0, stream>>>(Cql, q_left_deg, gcn_w_b, gcn_b, QNf, QNb, BQ, 0);
  neighbor_epilogue<<<(BQ * DD + T - 1) / T, T, 0, stream>>>(Cqr, q_right_deg, gcn_w_b, gcn_b, QNf, QNb, BQ, 1);

  // 4) support path (tiny) -> support_g
  support_neighbor_kernel<<<(BS * DM + T - 1) / T, T, 0, stream>>>(Ssl_f, Ssr_f, gcn_w, gcn_w_b, gcn_b,
                                                                   s_left_deg, s_right_deg, sn);
  support_mlp1_kernel<<<(BS * HH + T - 1) / T, T, 0, stream>>>(sn, p1_w, p1_b, h1);
  support_mlp2_kernel<<<(BS * DM + T - 1) / T, T, 0, stream>>>(h1, p2_w, p2_b, sn, ybuf);
  support_ln_mean_kernel<<<1, DM, 0, stream>>>(ybuf, ln_g, ln_b, sg);

  // 5) LSTM constants
  bias0_kernel<<<(FH + T - 1) / T, T, 0, stream>>>(b_ih, b_hh, bias0);
  gconst_kernel<<<(FH + T - 1) / T, T, 0, stream>>>(w_hh, sg, gc);

  // 6) qx = QN @ w_ih.T   [4096,512] x [4096,512]^T -> [4096,4096]
  {
    dim3 grid(BQ / 32, (FH / 64) / 8);
    wmma_gemm_bf16<<<grid, 256, 0, stream>>>(QNb, wih_b, qx, BQ, FH, DM);
  }

  // 7) LSTM steps: step 1 has h_r == 0 (skip recurrent GEMM, c_old = 0)
  lstm_step_kernel<<<(BQ * HH + T - 1) / T, T, 0, stream>>>(qx, Gm, bias0, gc, cbuf, QNf, hf, hb, 0);
  for (int s = 1; s < 4; ++s) {
    dim3 grid(BQ / 32, (FH / 64) / 8);
    wmma_gemm_bf16<<<grid, 256, 0, stream>>>(hb, whh_b, Gm, BQ, FH, DM);
    lstm_step_kernel<<<(BQ * HH + T - 1) / T, T, 0, stream>>>(qx, Gm, bias0, gc, cbuf, QNf, hf, hb, 1);
  }

  // 8) scores = h . support_g
  final_dot_kernel<<<BQ / 8, 256, 0, stream>>>(hf, sg, out);
}
